// Graph_Transformer_45681272160600
// MI455X (gfx1250) — matrix operand order repaired and checked by reading, not compiled, not run on hardware
//
#include <hip/hip_runtime.h>
#include <hip/hip_bf16.h>

typedef __bf16 bf16;
typedef __attribute__((ext_vector_type(16))) __bf16 v16bf;
typedef __attribute__((ext_vector_type(8)))  float  v8f;

// ---- problem constants (match reference) ----
constexpr int Bg  = 96;            // graphs
constexpr int NPG = 48;            // nodes per graph
constexpr int Hc  = 128;           // hidden
constexpr int NHc = 8;             // heads
constexpr int Lc  = 6;             // layers
constexpr int Nc  = Bg * NPG;      // 4608 nodes
constexpr int Ec  = 9216;          // bonds
constexpr int KMc = 32;            // mass RBF
constexpr int KDc = 64;            // dist RBF
constexpr int DHc = Hc / NHc;      // 16
constexpr int FFc = 4 * Hc;        // 512
constexpr int Rc  = Bg * NPG * NPG; // 221184 dense edge rows

// =====================================================================
// Weight prep: fp32 [K,N] row-major  ->  bf16 [N,K] (transposed)
// =====================================================================
__global__ void transpose_to_bf16_kernel(const float* __restrict__ src,
                                         bf16* __restrict__ dst, int K, int N) {
    int idx = blockIdx.x * blockDim.x + threadIdx.x;
    if (idx >= K * N) return;
    int k = idx / N, n = idx % N;
    dst[(size_t)n * K + k] = (bf16)src[(size_t)k * N + n];
}

__global__ void f32_to_bf16_kernel(const float* __restrict__ src,
                                   bf16* __restrict__ dst, size_t n) {
    size_t i = (size_t)blockIdx.x * blockDim.x + threadIdx.x;
    if (i < n) dst[i] = (bf16)src[i];
}

__global__ void copy_f32_kernel(const float* __restrict__ src,
                                float* __restrict__ dst, size_t n) {
    size_t i = (size_t)blockIdx.x * blockDim.x + threadIdx.x;
    if (i < n) dst[i] = src[i];
}

// =====================================================================
// Node embedding: x = sum of 9 categorical tables + massRBF @ mass_w + b
// one block per node, 128 threads (one per hidden channel)
// =====================================================================
__global__ void node_embed_kernel(const float* __restrict__ mass,
                                  const float* __restrict__ atom_emb,   // [9,16,H]
                                  const float* __restrict__ mass_w,     // [KM,H]
                                  const float* __restrict__ mass_b,     // [H]
                                  const int*   __restrict__ atom_feats, // [N,9]
                                  float* __restrict__ x, bf16* __restrict__ xbf) {
    int n = blockIdx.x;
    int h = threadIdx.x;
    __shared__ float rbfm[KMc];
    __shared__ int   feats[9];
    if (h < KMc) {
        float cm = 20.f * (float)h / (float)(KMc - 1);
        float dm = mass[n] - cm;
        rbfm[h] = __expf(-2.f * dm * dm);
    }
    if (h < 9) feats[h] = atom_feats[n * 9 + h];
    __syncthreads();
    float s = mass_b[h];
    #pragma unroll
    for (int f = 0; f < 9; ++f) s += atom_emb[((size_t)(f * 16 + feats[f])) * Hc + h];
    #pragma unroll 8
    for (int km = 0; km < KMc; ++km) s += rbfm[km] * mass_w[(size_t)km * Hc + h];
    size_t idx = (size_t)n * Hc + h;
    x[idx] = s;
    xbf[idx] = (bf16)s;
}

// =====================================================================
// Edge init: e[b,i,j,:] = RBF(d_ij)[64] @ rbf_w[64,128] + rbf_b
// Fused WMMA kernel: A fragments (16x32 bf16) are generated in-register
// from the distance RBF; B comes from pre-transposed bf16 rbf_w [128,64].
// One wave -> 16 rows x 128 cols. 8 waves / block.
// =====================================================================
__global__ void edge_rbf_kernel(const float* __restrict__ pos,     // [N,3]
                                const bf16*  __restrict__ rbfWt,   // [128,64] bf16
                                const float* __restrict__ rbf_b,   // [128]
                                float* __restrict__ e) {           // [R,128]
    int wave = blockIdx.x * (blockDim.x >> 5) + (threadIdx.x >> 5);
    int lane = threadIdx.x & 31;
    int r0 = wave * 16;
    if (r0 >= Rc) return;
    int mrow = lane & 15;
    int half = lane >> 4;

    // row -> (b,i,j) -> distance
    int p = r0 + mrow;
    int b = p / (NPG * NPG);
    int rem = p % (NPG * NPG);
    int i = rem / NPG, j = rem % NPG;
    const float* pi = pos + (size_t)(b * NPG + i) * 3;
    const float* pj = pos + (size_t)(b * NPG + j) * 3;
    float dx = pi[0] - pj[0], dy = pi[1] - pj[1], dz = pi[2] - pj[2];
    float d = sqrtf(dx * dx + dy * dy + dz * dz + 1e-12f);

    // Build A fragments (K=64 -> two 16x32 fragments), per ISA A layout:
    // element t (VGPR v=t/2, slot e=t%2): K = kb*32 + (v<4?0:16) + (v%4)*2 + half*8 + e
    v16bf afr[2];
    #pragma unroll
    for (int kb = 0; kb < 2; ++kb) {
        bf16* ap = (bf16*)&afr[kb];
        #pragma unroll
        for (int t = 0; t < 16; ++t) {
            int vv = t >> 1, ee = t & 1;
            int kk = kb * 32 + ((vv < 4) ? 0 : 16) + (vv & 3) * 2 + half * 8 + ee;
            float cd = 10.f * (float)kk / (float)(KDc - 1);
            float dd = d - cd;
            ap[t] = (bf16)__expf(-10.f * dd * dd);
        }
    }

    // Sweep 8 N-tiles of 16 columns
    for (int nb = 0; nb < 8; ++nb) {
        int ncol = nb * 16 + mrow;
        const bf16* wrow = rbfWt + (size_t)ncol * KDc;
        v8f acc = {0.f, 0.f, 0.f, 0.f, 0.f, 0.f, 0.f, 0.f};
        #pragma unroll
        for (int kb = 0; kb < 2; ++kb) {
            v16bf bm;
            uint4* bp = (uint4*)&bm;
            bp[0] = *(const uint4*)(wrow + kb * 32 + half * 8);
            bp[1] = *(const uint4*)(wrow + kb * 32 + 16 + half * 8);
            acc = __builtin_amdgcn_wmma_f32_16x16x32_bf16(
                false, afr[kb], false, bm, (short)0, acc, false, false);
        }
        float bias = rbf_b[ncol];
        #pragma unroll
        for (int vv = 0; vv < 8; ++vv) {
            size_t row = (size_t)(r0 + half * 8 + vv);
            e[row * Hc + ncol] = acc[vv] + bias;
        }
    }
}

// =====================================================================
// Bond scatter: e[gi,li,lj,:] += bond embedding (atomic, sparse E=9216)
// =====================================================================
__global__ void bond_scatter_kernel(const float* __restrict__ bond_emb,   // [3,8,H]
                                    const int*   __restrict__ bond_feats, // [E,3]
                                    const int*   __restrict__ edge_index, // [2,E]
                                    float* __restrict__ e) {
    int eid = blockIdx.x;
    int h = threadIdx.x;
    float s = 0.f;
    #pragma unroll
    for (int f = 0; f < 3; ++f) {
        int bf = bond_feats[eid * 3 + f];
        s += bond_emb[((size_t)(f * 8 + bf)) * Hc + h];
    }
    int src = edge_index[eid];
    int dst = edge_index[Ec + eid];
    int gi = src / NPG, li = src % NPG, lj = dst % NPG;
    atomicAdd(&e[(((size_t)gi * NPG + li) * NPG + lj) * Hc + h], s);
}

// =====================================================================
// Generic bf16 WMMA GEMM:  C[M,Nout] = act(A[M,K] @ W[K,Nout])
//   A    : bf16 row-major [M,K]
//   Wt   : bf16 TRANSPOSED weights [Nout,K]
//   Cf   : optional fp32 out, Cbf: optional bf16 out
// One wave computes 16 rows and sweeps all N tiles (A frags preloaded
// when K<=128). M must be a multiple of 16 (all call sites satisfy this).
// =====================================================================
__global__ void gemm_bf16_kernel(const bf16* __restrict__ A,
                                 const bf16* __restrict__ Wt,
                                 float* __restrict__ Cf,
                                 bf16*  __restrict__ Cbf,
                                 int M, int K, int Nout, int relu) {
    int wave = blockIdx.x * (blockDim.x >> 5) + (threadIdx.x >> 5);
    int lane = threadIdx.x & 31;
    int r0 = wave * 16;
    if (r0 >= M) return;
    int mrow = lane & 15;
    int half = lane >> 4;
    int nTiles = (Nout + 15) >> 4;
    int kTiles = K >> 5;

    const bf16* arow = A + (size_t)(r0 + mrow) * K;

    v16bf afr[4];
    bool pre = (kTiles <= 4);
    if (pre) {
        for (int kb = 0; kb < kTiles; ++kb) {
            int k0 = kb * 32;
            uint4* ap = (uint4*)&afr[kb];
            ap[0] = *(const uint4*)(arow + k0 + half * 8);
            ap[1] = *(const uint4*)(arow + k0 + 16 + half * 8);
        }
    }

    for (int nb = 0; nb < nTiles; ++nb) {
        int ncol = nb * 16 + mrow;
        bool nvalid = (ncol < Nout);
        const bf16* wrow = Wt + (size_t)ncol * K;
        v8f acc = {0.f, 0.f, 0.f, 0.f, 0.f, 0.f, 0.f, 0.f};
        for (int kb = 0; kb < kTiles; ++kb) {
            int k0 = kb * 32;
            v16bf a;
            if (pre) {
                a = afr[kb & 3];
            } else {
                uint4* ap = (uint4*)&a;
                ap[0] = *(const uint4*)(arow + k0 + half * 8);
                ap[1] = *(const uint4*)(arow + k0 + 16 + half * 8);
            }
            v16bf bm;
            uint4* bp = (uint4*)&bm;
            if (nvalid) {
                bp[0] = *(const uint4*)(wrow + k0 + half * 8);
                bp[1] = *(const uint4*)(wrow + k0 + 16 + half * 8);
            } else {
                uint4 z = make_uint4(0u, 0u, 0u, 0u);
                bp[0] = z; bp[1] = z;
            }
            acc = __builtin_amdgcn_wmma_f32_16x16x32_bf16(
                false, a, false, bm, (short)0, acc, false, false);
        }
        if (relu) {
            #pragma unroll
            for (int vv = 0; vv < 8; ++vv) acc[vv] = fmaxf(acc[vv], 0.f);
        }
        if (nvalid) {
            #pragma unroll
            for (int vv = 0; vv < 8; ++vv) {
                size_t idx = (size_t)(r0 + half * 8 + vv) * Nout + ncol;
                if (Cf)  Cf[idx]  = acc[vv];
                if (Cbf) Cbf[idx] = (bf16)acc[vv];
            }
        }
    }
}

// =====================================================================
// Attention (scalar; tiny FLOPs): per (b,i,h): softmax over j=0..47
// of scale*q.k + bias, then o = sum a_j v_j. Writes bf16 o (Wo GEMM in).
// =====================================================================
__global__ void attention_kernel(const float* __restrict__ q,
                                 const float* __restrict__ k,
                                 const float* __restrict__ v,
                                 const float* __restrict__ bias,  // [R, NH]
                                 bf16* __restrict__ obf) {
    int t = blockIdx.x * blockDim.x + threadIdx.x;
    if (t >= Bg * NPG * NHc) return;
    int h  = t % NHc;
    int bi = t / NHc;
    int i  = bi % NPG;
    int b  = bi / NPG;

    const float* qp = q + (size_t)(b * NPG + i) * Hc + h * DHc;
    float qr[DHc];
    #pragma unroll
    for (int d = 0; d < DHc; ++d) qr[d] = qp[d];

    const float scale = 0.25f;  // DH^-0.5, DH=16
    float s[NPG];
    float mx = -3.402823466e38f;
    size_t rowbase = (size_t)b * (NPG * NPG) + (size_t)i * NPG;
    for (int j = 0; j < NPG; ++j) {
        const float* kp = k + (size_t)(b * NPG + j) * Hc + h * DHc;
        float acc = 0.f;
        #pragma unroll
        for (int d = 0; d < DHc; ++d) acc += qr[d] * kp[d];
        float sv = acc * scale + bias[(rowbase + j) * NHc + h];
        s[j] = sv;
        mx = fmaxf(mx, sv);
    }
    float sum = 0.f;
    for (int j = 0; j < NPG; ++j) { float ev = __expf(s[j] - mx); s[j] = ev; sum += ev; }
    float inv = 1.f / sum;
    float o[DHc];
    #pragma unroll
    for (int d = 0; d < DHc; ++d) o[d] = 0.f;
    for (int j = 0; j < NPG; ++j) {
        float a = s[j] * inv;
        const float* vp = v + (size_t)(b * NPG + j) * Hc + h * DHc;
        #pragma unroll
        for (int d = 0; d < DHc; ++d) o[d] += a * vp[d];
    }
    bf16* op = obf + (size_t)(b * NPG + i) * Hc + h * DHc;
    #pragma unroll
    for (int d = 0; d < DHc; ++d) op[d] = (bf16)o[d];
}

// =====================================================================
// Residual + LayerNorm over H=128: out = LN(base + delta), also bf16 copy
// one block (128 threads) per row; LDS tree reductions.
// =====================================================================
__global__ void ln_residual_kernel(const float* __restrict__ base,
                                   const float* __restrict__ delta,
                                   float* __restrict__ out_f,
                                   bf16*  __restrict__ out_bf, int M) {
    int r = blockIdx.x;
    if (r >= M) return;
    int t = threadIdx.x;
    size_t idx = (size_t)r * Hc + t;
    float v = base[idx] + delta[idx];
    __shared__ float red[Hc];
    red[t] = v;
    __syncthreads();
    for (int s = Hc / 2; s > 0; s >>= 1) {
        if (t < s) red[t] += red[t + s];
        __syncthreads();
    }
    float mean = red[0] * (1.f / Hc);
    __syncthreads();
    float c = v - mean;
    red[t] = c * c;
    __syncthreads();
    for (int s = Hc / 2; s > 0; s >>= 1) {
        if (t < s) red[t] += red[t + s];
        __syncthreads();
    }
    float rstd = rsqrtf(red[0] * (1.f / Hc) + 1e-5f);
    float y = c * rstd;
    out_f[idx] = y;
    if (out_bf) out_bf[idx] = (bf16)y;
}

// =====================================================================
// Host: carve workspace, prep weights, run pipeline
// =====================================================================
extern "C" void kernel_launch(void* const* d_in, const int* in_sizes, int n_in,
                              void* d_out, int out_size, void* d_ws, size_t ws_size,
                              hipStream_t stream) {
    (void)in_sizes; (void)n_in; (void)out_size; (void)ws_size;

    const float* mass       = (const float*)d_in[0];
    const float* pos        = (const float*)d_in[1];
    const float* atom_emb   = (const float*)d_in[2];
    const float* mass_w     = (const float*)d_in[3];
    const float* mass_b     = (const float*)d_in[4];
    const float* bond_emb   = (const float*)d_in[5];
    const float* rbf_w      = (const float*)d_in[6];
    const float* rbf_b      = (const float*)d_in[7];
    const float* Wq         = (const float*)d_in[8];
    const float* Wk         = (const float*)d_in[9];
    const float* Wv         = (const float*)d_in[10];
    const float* Wo         = (const float*)d_in[11];
    const float* Wb         = (const float*)d_in[12];
    const float* W1         = (const float*)d_in[13];
    const float* W2         = (const float*)d_in[14];
    const float* We1        = (const float*)d_in[15];
    const float* We2        = (const float*)d_in[16];
    const int*   atom_feats = (const int*)d_in[17];
    const int*   bond_feats = (const int*)d_in[18];
    const int*   edge_index = (const int*)d_in[19];

    // ---- workspace carve ----
    char* wsp = (char*)d_ws;
    auto alloc = [&](size_t bytes) -> char* {
        char* r = wsp;
        wsp += (bytes + 255) & ~(size_t)255;
        return r;
    };
    float* e     = (float*)alloc((size_t)Rc * Hc * 4);   // fp32 edge state (113MB, L2-resident)
    bf16*  ebf   = (bf16*)alloc((size_t)Rc * Hc * 2);    // bf16 mirror (WMMA A input)
    float* etmp  = (float*)alloc((size_t)Rc * Hc * 4);   // edge FFN output pre-LN
    bf16*  ehid  = (bf16*)alloc((size_t)Rc * Hc * 2);    // relu(e@We1) bf16
    float* bias  = (float*)alloc((size_t)Rc * NHc * 4);  // attention bias [R,8]
    float* x     = (float*)alloc((size_t)Nc * Hc * 4);
    bf16*  xbf   = (bf16*)alloc((size_t)Nc * Hc * 2);
    float* qb    = (float*)alloc((size_t)Nc * Hc * 4);
    float* kb    = (float*)alloc((size_t)Nc * Hc * 4);
    float* vb    = (float*)alloc((size_t)Nc * Hc * 4);
    bf16*  obf   = (bf16*)alloc((size_t)Nc * Hc * 2);
    float* xtmp  = (float*)alloc((size_t)Nc * Hc * 4);
    bf16*  hbf   = (bf16*)alloc((size_t)Nc * FFc * 2);   // node FFN hidden bf16

    // bf16 transposed weights
    bf16* rbfWt = (bf16*)alloc((size_t)Hc * KDc * 2);
    bf16 *WqT[Lc], *WkT[Lc], *WvT[Lc], *WoT[Lc], *WbT[Lc], *W1T[Lc], *W2T[Lc], *We1T[Lc], *We2T[Lc];
    for (int l = 0; l < Lc; ++l) {
        WqT[l]  = (bf16*)alloc((size_t)Hc * Hc * 2);
        WkT[l]  = (bf16*)alloc((size_t)Hc * Hc * 2);
        WvT[l]  = (bf16*)alloc((size_t)Hc * Hc * 2);
        WoT[l]  = (bf16*)alloc((size_t)Hc * Hc * 2);
        WbT[l]  = (bf16*)alloc((size_t)NHc * Hc * 2);
        W1T[l]  = (bf16*)alloc((size_t)FFc * Hc * 2);
        W2T[l]  = (bf16*)alloc((size_t)Hc * FFc * 2);
        We1T[l] = (bf16*)alloc((size_t)Hc * Hc * 2);
        We2T[l] = (bf16*)alloc((size_t)Hc * Hc * 2);
    }

    auto tp = [&](const float* src, bf16* dst, int K, int Nn) {
        int n = K * Nn;
        transpose_to_bf16_kernel<<<(n + 255) / 256, 256, 0, stream>>>(src, dst, K, Nn);
    };
    tp(rbf_w, rbfWt, KDc, Hc);
    for (int l = 0; l < Lc; ++l) {
        tp(Wq  + (size_t)l * Hc * Hc,  WqT[l],  Hc,  Hc);
        tp(Wk  + (size_t)l * Hc * Hc,  WkT[l],  Hc,  Hc);
        tp(Wv  + (size_t)l * Hc * Hc,  WvT[l],  Hc,  Hc);
        tp(Wo  + (size_t)l * Hc * Hc,  WoT[l],  Hc,  Hc);
        tp(Wb  + (size_t)l * Hc * NHc, WbT[l],  Hc,  NHc);
        tp(W1  + (size_t)l * Hc * FFc, W1T[l],  Hc,  FFc);
        tp(W2  + (size_t)l * FFc * Hc, W2T[l],  FFc, Hc);
        tp(We1 + (size_t)l * Hc * Hc,  We1T[l], Hc,  Hc);
        tp(We2 + (size_t)l * Hc * Hc,  We2T[l], Hc,  Hc);
    }

    auto gemm = [&](const bf16* A, const bf16* Wt, float* Cf, bf16* Cbf,
                    int M, int K, int Nout, int relu) {
        int waves = M / 16;
        int blocks = (waves + 7) / 8;
        gemm_bf16_kernel<<<blocks, 256, 0, stream>>>(A, Wt, Cf, Cbf, M, K, Nout, relu);
    };

    // ---- embeddings ----
    node_embed_kernel<<<Nc, Hc, 0, stream>>>(mass, atom_emb, mass_w, mass_b, atom_feats, x, xbf);
    edge_rbf_kernel<<<Rc / 128, 256, 0, stream>>>(pos, rbfWt, rbf_b, e);
    bond_scatter_kernel<<<Ec, Hc, 0, stream>>>(bond_emb, bond_feats, edge_index, e);
    {
        size_t n = (size_t)Rc * Hc;
        f32_to_bf16_kernel<<<(unsigned)((n + 255) / 256), 256, 0, stream>>>(e, ebf, n);
    }

    // ---- transformer layers ----
    for (int l = 0; l < Lc; ++l) {
        // projections (bf16 WMMA, f32 out for scalar attention)
        gemm(xbf, WqT[l], qb, nullptr, Nc, Hc, Hc, 0);
        gemm(xbf, WkT[l], kb, nullptr, Nc, Hc, Hc, 0);
        gemm(xbf, WvT[l], vb, nullptr, Nc, Hc, Hc, 0);
        // edge-bias projection: [R,128] x [128,8]
        gemm(ebf, WbT[l], bias, nullptr, Rc, Hc, NHc, 0);
        // attention + softmax (scalar; tiny)
        attention_kernel<<<(Bg * NPG * NHc) / 128, 128, 0, stream>>>(qb, kb, vb, bias, obf);
        // out projection + residual LN
        gemm(obf, WoT[l], xtmp, nullptr, Nc, Hc, Hc, 0);
        ln_residual_kernel<<<Nc, Hc, 0, stream>>>(x, xtmp, x, xbf, Nc);
        // node FFN 128 -> 512 -> 128
        gemm(xbf, W1T[l], nullptr, hbf, Nc, Hc, FFc, 1);
        gemm(hbf, W2T[l], xtmp, nullptr, Nc, FFc, Hc, 0);
        ln_residual_kernel<<<Nc, Hc, 0, stream>>>(x, xtmp, x, xbf, Nc);
        // edge FFN 128 -> 128 -> 128 (dominant GEMMs)
        gemm(ebf, We1T[l], nullptr, ehid, Rc, Hc, Hc, 1);
        gemm(ehid, We2T[l], etmp, nullptr, Rc, Hc, Hc, 0);
        ln_residual_kernel<<<Rc, Hc, 0, stream>>>(e, etmp, e, ebf, Rc);
    }

    // ---- output: x [N,H] fp32 ----
    {
        size_t n = (size_t)Nc * Hc;
        copy_f32_kernel<<<(unsigned)((n + 255) / 256), 256, 0, stream>>>(x, (float*)d_out, n);
    }
}
